// BaseEncoderBiRNN_73976516706549
// MI455X (gfx1250) — compile-verified, matrix operand order checked by reading, not run on hardware
//
#include <hip/hip_runtime.h>
#include <stdint.h>

// Bidirectional LSTM, H=1024, L=2048, batch 1.
// Phase 1: gx = x @ W_ih^T + b_ih + b_hh  (bf16 WMMA, f32 accumulate, 16x64/wave)
// Phase 2: persistent recurrent kernel; W_hh slice resident in LDS as f32,
//          preloaded via TDM (tensor_load_to_lds), h staged per-step via
//          global_load_async_to_lds_b128, one grid-sync per timestep/direction.

#define HD   1024
#define LSEQ 2048
#define G4H  4096
#define NWG  64     // workgroups per direction in the persistent kernel
#define TPB  256

typedef __attribute__((ext_vector_type(16))) __bf16   v16bf;
typedef __attribute__((ext_vector_type(8)))  float    v8f;
typedef __attribute__((ext_vector_type(4)))  uint32_t v4u;
typedef __attribute__((ext_vector_type(8)))  uint32_t v8u;

union FragU { v16bf v; uint32_t u[8]; };

__device__ inline uint16_t f2bf(float f) {
    uint32_t u = __float_as_uint(f);
    uint32_t r = (u + 0x7FFFu + ((u >> 16) & 1u)) >> 16;   // round-to-nearest-even
    return (uint16_t)r;
}

// ---------------------------------------------------------------- init / convert

__global__ void init_ws(float* __restrict__ hbuf, int* __restrict__ cnt) {
    int i = threadIdx.x + blockIdx.x * blockDim.x;
    if (i < 2 * HD) hbuf[i] = 0.0f;
    if (i < 2) cnt[i] = 0;
}

__global__ void convert_bf16(const float* __restrict__ in, uint16_t* __restrict__ out, int n) {
    int i = threadIdx.x + blockIdx.x * blockDim.x;
    int stride = blockDim.x * gridDim.x;
    for (; i < n; i += stride) out[i] = f2bf(in[i]);
}

// ---------------------------------------------------------------- WMMA GEMM: gx

// 16-bit A-matrix 16x32 fragment load (ISA 7.12.2 layout).
// lanes 0-15: M=lane, K in {0..7, 16..23}; lanes 16-31: M=lane-16, K in {8..15, 24..31}
__device__ inline v16bf load_frag16(const uint16_t* rowbase /* already + kb */, int k0) {
    FragU f;
    const uint4* p = (const uint4*)(rowbase + k0);   // 8 bf16 = 16B
    uint4 lo = p[0];       // K = kb+0..kb+7
    uint4 hi = p[2];       // K = kb+16..kb+23
    f.u[0] = lo.x; f.u[1] = lo.y; f.u[2] = lo.z; f.u[3] = lo.w;
    f.u[4] = hi.x; f.u[5] = hi.y; f.u[6] = hi.z; f.u[7] = hi.w;
    return f.v;
}

__global__ __launch_bounds__(TPB)
void gemm_gx(const uint16_t* __restrict__ xbf,
             const uint16_t* __restrict__ wf, const uint16_t* __restrict__ wr,
             const float* __restrict__ bihf, const float* __restrict__ bhhf,
             const float* __restrict__ bihr, const float* __restrict__ bhhr,
             float* __restrict__ gx0, float* __restrict__ gx1) {
    const int d = blockIdx.y;
    const uint16_t* wb = d ? wr : wf;
    const float* b1 = d ? bihr : bihf;
    const float* b2 = d ? bhhr : bhhf;
    float* gx = d ? gx1 : gx0;

    const int lane  = threadIdx.x & 31;
    const int wave  = threadIdx.x >> 5;
    const int mtile = blockIdx.x >> 3;            // 128 M-tiles (time)
    const int ngrp  = blockIdx.x & 7;             // 8 groups; wave covers 64 N cols
    const int m0 = mtile * 16;
    const int n0 = (ngrp * 8 + wave) * 64;        // 4 consecutive 16-wide N tiles

    const int mr = lane & 15;
    const int kb = (lane >> 4) << 3;              // 0 or 8

    const uint16_t* arow  = xbf + (size_t)(m0 + mr) * HD + kb;  // A: x tile (M x K)
    const uint16_t* brow0 = wb  + (size_t)(n0 + mr) * HD + kb;  // B: W rows (N x K)

    v8f acc[4] = {v8f{}, v8f{}, v8f{}, v8f{}};
    for (int k0 = 0; k0 < HD; k0 += 32) {
        __builtin_prefetch(arow + k0 + 256, 0, 1);              // global_prefetch_b8
        v16bf a = load_frag16(arow, k0);
#pragma unroll
        for (int j = 0; j < 4; ++j) {
            __builtin_prefetch(brow0 + (size_t)j * 16 * HD + k0 + 256, 0, 1);
            v16bf b = load_frag16(brow0 + (size_t)j * 16 * HD, k0);
            acc[j] = __builtin_amdgcn_wmma_f32_16x16x32_bf16(
                         /*neg_a=*/false, a, /*neg_b=*/false, b,
                         /*c_mod=*/(short)0, acc[j], /*reuse_a=*/false, /*reuse_b=*/false);
        }
    }

    // C/D layout: lane 0-15 -> M=0..7 (vgpr r), lane 16-31 -> M=8..15; N=lane&15
    const int nc = lane & 15;
    const int mb = (lane >> 4) << 3;
#pragma unroll
    for (int j = 0; j < 4; ++j) {
        const int nj = n0 + j * 16;
        const float bias = b1[nj + nc] + b2[nj + nc];
        float* orow = gx + (size_t)(m0 + mb) * G4H + nj + nc;
#pragma unroll
        for (int r = 0; r < 8; ++r)
            orow[(size_t)r * G4H] = acc[j][r] + bias;
    }
}

// ---------------------------------------------------------------- persistent scan

__global__ __launch_bounds__(TPB)
void lstm_scan(const float* __restrict__ whhf, const float* __restrict__ whhr,
               const float* __restrict__ gx0, const float* __restrict__ gx1,
               float* __restrict__ hbuf /*2*H*/, int* __restrict__ cnt /*2*/,
               float* __restrict__ out) {
    extern __shared__ unsigned char smem[];
    float* wlds  = (float*)smem;                        // 64 rows x 1024 f32 = 256 KB
    float* hsh   = (float*)(smem + (size_t)64 * HD * 4); // 4 KB
    float* gates = hsh + HD;                            // 64
    float* cst   = gates + 64;                          // 16 (persistent cell state)

    const int tid = threadIdx.x;
    const int d   = blockIdx.x >> 6;          // direction
    const int wg  = blockIdx.x & 63;          // slice within direction
    const int u0  = wg * 16;                  // first hidden unit of this WG
    const float* whh = d ? whhr : whhf;
    const float* gx  = d ? gx1  : gx0;
    float* hb  = hbuf + d * HD;
    int* mycnt = cnt + d;

    // --- W_hh slice -> LDS via Tensor Data Mover: 4 descriptors, one per gate
    // block; each is a 16x1024 f32 2D tile (row stride 1024) of W_hh starting at
    // row gt*H + u0.  D# layout per CDNA5 ISA.
    if (tid < 32) {                           // wave 0 issues; EXEC ignored by TDM
        const uint32_t lds_base = (uint32_t)(uintptr_t)wlds;
        for (int gt = 0; gt < 4; ++gt) {
            const uint64_t ga = (uint64_t)(uintptr_t)(whh + (size_t)(gt * HD + u0) * HD);
            v4u g0; v8u g1;
            g0[0] = 1u;                                            // count=1, user desc
            g0[1] = lds_base + (uint32_t)gt * 16u * HD * 4u;       // lds_addr (bytes)
            g0[2] = (uint32_t)ga;                                  // global_addr[31:0]
            g0[3] = (uint32_t)((ga >> 32) & 0x01FFFFFFu) | 0x80000000u; // ga[56:32] | type=2
            g1[0] = 0x00020000u;              // wg_mask=0, data_size=2 (4B)
            g1[1] = (1024u & 0xFFFFu) << 16;  // tensor_dim0[15:0] in [63:48]
            g1[2] = (1024u >> 16) | (16u << 16);   // tensor_dim0[31:16] | tensor_dim1[15:0]
            g1[3] = (16u >> 16) | (1024u << 16);   // tensor_dim1[31:16] | tile_dim0=1024
            g1[4] = 16u;                      // tile_dim1=16, tile_dim2=0
            g1[5] = 1024u;                    // tensor_dim0_stride low32
            g1[6] = 0u;                       // stride0 hi | stride1 lo
            g1[7] = 0u;
            asm volatile("tensor_load_to_lds %0, %1" :: "s"(g0), "s"(g1) : "memory");
        }
        __builtin_amdgcn_s_wait_tensorcnt(0);
    }
    if (tid < 16) cst[tid] = 0.0f;
    __syncthreads();

    const int row = tid >> 2;                 // 64 gate rows per WG
    const int q   = tid & 3;                  // quarter of K
    const int gt  = row >> 4, lu = row & 15;
    const float* wrow = wlds + (size_t)row * HD + q * (HD / 4);

    for (int s = 0; s < LSEQ; ++s) {
        const int trow = d ? (LSEQ - 1 - s) : s;   // gx row & output row

        // stage h vector (4KB) into LDS: one async B128 per lane, 8 waves total
        {
            const uint32_t dst = (uint32_t)(uintptr_t)hsh + (uint32_t)tid * 16u;
            const float* src = hb + tid * 4;
            asm volatile("global_load_async_to_lds_b128 %0, %1, off"
                         :: "v"(dst), "v"(src) : "memory");
            asm volatile("s_wait_asynccnt 0x0" ::: "memory");
        }
        __syncthreads();

        float gxv = 0.0f;
        if (q == 0) gxv = gx[(size_t)trow * G4H + gt * HD + u0 + lu];

        const float* hq = hsh + q * (HD / 4);
        float sum = 0.0f;
#pragma unroll 4
        for (int i = 0; i < HD / 16; ++i) {           // 64 iters of float4 x float4
            const float4 w4 = ((const float4*)wrow)[i];
            const float4 h4 = ((const float4*)hq)[i];
            sum = fmaf(w4.x, h4.x, sum);
            sum = fmaf(w4.y, h4.y, sum);
            sum = fmaf(w4.z, h4.z, sum);
            sum = fmaf(w4.w, h4.w, sum);
        }
        sum += __shfl_xor(sum, 1, 32);
        sum += __shfl_xor(sum, 2, 32);
        if (q == 0) gates[row] = gxv + sum;
        __syncthreads();

        if (tid < 16) {
            float iv = gates[tid];
            float fv = gates[16 + tid];
            float gv = gates[32 + tid];
            float ov = gates[48 + tid];
            iv = 1.0f / (1.0f + __expf(-iv));
            fv = 1.0f / (1.0f + __expf(-fv));
            gv = tanhf(gv);
            ov = 1.0f / (1.0f + __expf(-ov));
            float cn = fv * cst[tid] + iv * gv;
            cst[tid] = cn;
            float h = ov * tanhf(cn);
            int gu = u0 + tid;
            hb[gu] = h;
            out[(size_t)trow * (2 * HD) + d * HD + gu] = h;
            if (s == LSEQ - 1) {
                size_t base = (size_t)LSEQ * 2 * HD;
                int col = d * HD + gu;
                out[base + col]          = h;    // hidden (copy 1)
                out[base + 2 * HD + col] = cn;   // cell   (copy 1)
                out[base + 4 * HD + col] = h;    // hidden (copy 2)
                out[base + 6 * HD + col] = cn;   // cell   (copy 2)
            }
        }
        if (tid == 0) {   // per-direction grid barrier (monotonic counter, zeroed per launch)
            __threadfence();
            __hip_atomic_fetch_add(mycnt, 1, __ATOMIC_RELEASE, __HIP_MEMORY_SCOPE_AGENT);
            const int target = NWG * (s + 1);
            while (__hip_atomic_load(mycnt, __ATOMIC_ACQUIRE, __HIP_MEMORY_SCOPE_AGENT) < target)
                __builtin_amdgcn_s_sleep(8);
            __threadfence();
        }
        __syncthreads();
    }
}

// ---------------------------------------------------------------- launch

extern "C" void kernel_launch(void* const* d_in, const int* in_sizes, int n_in,
                              void* d_out, int out_size, void* d_ws, size_t ws_size,
                              hipStream_t stream) {
    const float* x    = (const float*)d_in[0];
    const float* Wihf = (const float*)d_in[1];
    const float* Whhf = (const float*)d_in[2];
    const float* bihf = (const float*)d_in[3];
    const float* bhhf = (const float*)d_in[4];
    const float* Wihr = (const float*)d_in[5];
    const float* Whhr = (const float*)d_in[6];
    const float* bihr = (const float*)d_in[7];
    const float* bhhr = (const float*)d_in[8];
    float* out = (float*)d_out;

    // workspace layout
    uint16_t* xbf = (uint16_t*)d_ws;                       // L*H bf16
    uint16_t* wfb = xbf + (size_t)LSEQ * HD;               // 4H*H bf16
    uint16_t* wrb = wfb + (size_t)G4H * HD;                // 4H*H bf16
    float* gx0  = (float*)(wrb + (size_t)G4H * HD);        // L*4H f32
    float* gx1  = gx0 + (size_t)LSEQ * G4H;                // L*4H f32
    float* hbuf = gx1 + (size_t)LSEQ * G4H;                // 2*H f32
    int*   cnt  = (int*)(hbuf + 2 * HD);                   // 2 ints

    init_ws<<<dim3(8), dim3(256), 0, stream>>>(hbuf, cnt);
    convert_bf16<<<dim3(512), 256, 0, stream>>>(x,    xbf, LSEQ * HD);
    convert_bf16<<<dim3(1024), 256, 0, stream>>>(Wihf, wfb, G4H * HD);
    convert_bf16<<<dim3(1024), 256, 0, stream>>>(Wihr, wrb, G4H * HD);

    // 128 M-tiles x 8 N-groups per direction; 8 waves/block, 16x64 strip per wave
    gemm_gx<<<dim3(1024, 2), TPB, 0, stream>>>(xbf, wfb, wrb,
                                               bihf, bhhf, bihr, bhhr, gx0, gx1);

    size_t smem = (size_t)64 * HD * sizeof(float)      // W_hh slice (256 KB, f32)
                + HD * sizeof(float)                   // h cache
                + 64 * sizeof(float)                   // gates
                + 16 * sizeof(float);                  // cell state
    lstm_scan<<<dim3(2 * NWG), TPB, smem, stream>>>(Whhf, Whhr, gx0, gx1, hbuf, cnt, out);
}
